// GCNMappingNetwork_54425825575565
// MI455X (gfx1250) — matrix operand order repaired and checked by reading, not compiled, not run on hardware
//
#include <hip/hip_runtime.h>
#include <hip/hip_bf16.h>
#include <math.h>

// ---------------- problem constants ----------------
#define B_      16
#define N_      250
#define BN      4000        // B_ * N_          (real rows)
#define NP      256         // padded rows per graph
#define BNP     4096        // B_ * NP          (padded rows)
#define ZD      16
#define PD      16
#define IN0     32          // ZD + PD
#define WD      512
#define NUM_WS  14
#define LR_     0.01f
#define SQRT2_  1.41421356237309505f

typedef __bf16 bf16_t;
typedef __attribute__((ext_vector_type(8)))  __bf16 v8bf;
typedef __attribute__((ext_vector_type(16))) __bf16 v16bf;
typedef __attribute__((ext_vector_type(8)))  float  v8f;

union V16bf { v16bf v; v8bf h[2]; };

// ---------------- prep kernels (one-time per call, elementwise) ----------------

// inv[rp] = 1/rowsum for real rows, 0 for pad rows (so agg stores are unguarded:
// pad rows become exact zeros). One wave per padded row; only the diagonal
// block of adj can be non-zero, so sum 250 columns.
__global__ void rowsum_kernel(const float* __restrict__ adj, float* __restrict__ inv) {
    int wave = threadIdx.x >> 5;
    int lane = threadIdx.x & 31;
    int rp   = blockIdx.x * 8 + wave;          // 0..4095
    if (rp >= BNP) return;
    int g = rp >> 8, i = rp & (NP - 1);
    if (i < N_) {
        const float* p = adj + (size_t)(g * N_ + i) * BN + (size_t)g * N_;
        float s = 0.f;
        for (int c = lane; c < N_; c += 32) s += p[c];
        #pragma unroll
        for (int o = 16; o > 0; o >>= 1) s += __shfl_xor(s, o, 32);
        if (lane == 0) inv[rp] = 1.0f / s;
    } else if (lane == 0) {
        inv[rp] = 0.0f;
    }
}

// Zero-padded bf16 copy of the 16 diagonal blocks: adjp[g][i][k], 16x256x256.
// Binary values are exact in bf16; makes the hot agg loop fully branchless.
__global__ void adjp_kernel(const float* __restrict__ adj, bf16_t* __restrict__ adjp) {
    int t = blockIdx.x * blockDim.x + threadIdx.x;     // 16*256*256 = 1M
    if (t >= B_ * NP * NP) return;
    int k = t & (NP - 1), i = (t >> 8) & (NP - 1), g = t >> 16;
    float v = 0.f;
    if (i < N_ && k < N_) v = adj[(size_t)(g * N_ + i) * BN + (size_t)g * N_ + k];
    adjp[t] = (bf16_t)v;
}

// x0 transposed (it is consumed as a B operand): x0T[d][rp], d in 0..31,
// pad rows zero. Row rp -> real row r = g*250+i, z row = r % 16.
__global__ void build_x0_kernel(const float* __restrict__ z, const float* __restrict__ poses,
                                bf16_t* __restrict__ x0T) {
    int t = blockIdx.x * blockDim.x + threadIdx.x;     // 4096*32
    if (t >= BNP * IN0) return;
    int rp = t & (BNP - 1), d = t >> 12;
    int g = rp >> 8, i = rp & (NP - 1);
    float v = 0.f;
    if (i < N_) {
        int r = g * N_ + i;
        if (d < ZD) {
            int zr = r & (B_ - 1);
            float ss = 0.f;
            #pragma unroll
            for (int k = 0; k < ZD; ++k) { float zz = z[zr * ZD + k]; ss += zz * zz; }
            v = z[zr * ZD + d] * __frsqrt_rn(ss * (1.0f / ZD) + 1e-8f);
        } else {
            v = poses[r * PD + (d - ZD)];
        }
    }
    x0T[(size_t)d * BNP + rp] = (bf16_t)v;
}

// Weights: the FC B-operand wants w[n][k] -- exactly the input layout.
// So this is a straight flat f32 -> bf16 convert, no transpose.
__global__ void wprep_kernel(const float* __restrict__ w_first, const float* __restrict__ w_rest,
                             bf16_t* __restrict__ wt) {
    const int L0  = IN0 * WD;
    const int TOT = L0 + 7 * WD * WD;
    int t = blockIdx.x * blockDim.x + threadIdx.x;
    if (t >= TOT) return;
    wt[t] = (bf16_t)(t < L0 ? w_first[t] : w_rest[t - L0]);
}

// ---------------- WMMA GEMM kernels (branchless, b128 fragment loads) ----------------

#define WMMA_BF16(A, Bm, C) \
    __builtin_amdgcn_wmma_f32_16x16x32_bf16(false, (A), false, (Bm), (short)0, (C), false, false)

// y[rp][n] = inv[rp] * sum_k adjp[g][i][k] * hT[n][g*256+k]
// One wave per 16x16 M-tile x two N-tiles (A fragment reused).
__global__ __launch_bounds__(32)
void agg_kernel(const bf16_t* __restrict__ adjp, const float* __restrict__ inv,
                const bf16_t* __restrict__ hT, bf16_t* __restrict__ y, int W) {
    int lane  = threadIdx.x;
    int nPair = blockIdx.x;                 // W/32
    int mTile = blockIdx.y;                 // 16
    int g     = blockIdx.z;                 // 16
    int hi = lane >> 4, col = lane & 15;
    int row = mTile * 16 + col;
    int n0 = nPair * 32 + col, n1 = n0 + 16;
    const bf16_t* aB  = adjp + (size_t)(g * NP + row) * NP;
    const bf16_t* b0B = hT + (size_t)n0 * BNP + g * NP;
    const bf16_t* b1B = hT + (size_t)n1 * BNP + g * NP;
    v8f acc0 = {}, acc1 = {};
    #pragma unroll
    for (int kk = 0; kk < NP; kk += 32) {
        V16bf a, b0, b1;
        a.h[0]  = *(const v8bf*)(aB + kk + hi * 16);         // A row, K = kk+hi*16 .. +7
        a.h[1]  = *(const v8bf*)(aB + kk + hi * 16 + 8);     // A row, K = kk+hi*16+8 .. +15
        b0.h[0] = *(const v8bf*)(b0B + kk + hi * 16);        // B col, K = kk+hi*16 .. +15
        b0.h[1] = *(const v8bf*)(b0B + kk + hi * 16 + 8);
        b1.h[0] = *(const v8bf*)(b1B + kk + hi * 16);
        b1.h[1] = *(const v8bf*)(b1B + kk + hi * 16 + 8);
        acc0 = WMMA_BF16(a.v, b0.v, acc0);
        acc1 = WMMA_BF16(a.v, b1.v, acc1);
    }
    #pragma unroll
    for (int r = 0; r < 8; ++r) {                            // D: VGPR r -> row M = hi*8+r
        int m = mTile * 16 + hi * 8 + r;
        float s = inv[g * NP + m];                           // 0 on pad rows -> exact zero
        size_t rowOff = (size_t)(g * NP + m) * W;
        y[rowOff + n0] = (bf16_t)(acc0[r] * s);
        y[rowOff + n1] = (bf16_t)(acc1[r] * s);
    }
}

// hT[n][rp] = lrelu( (y @ w^T)[rp][n] * wg + bias[n]*LR ) * sqrt(2)
// A from row-major y (b128), B from wt[n][k] (b128), D written transposed:
// each lane's 8 output rows are contiguous -> one 16B store per N-tile.
__global__ __launch_bounds__(32)
void fc_kernel(const bf16_t* __restrict__ y, const bf16_t* __restrict__ wt,
               const float* __restrict__ bias, bf16_t* __restrict__ hT,
               int IN, float wg) {
    int lane  = threadIdx.x;
    int nPair = blockIdx.x;                 // 16
    int mTile = blockIdx.y;                 // 256
    int hi = lane >> 4, col = lane & 15;
    int row = mTile * 16 + col;
    int n0 = nPair * 32 + col, n1 = n0 + 16;
    const bf16_t* aB  = y  + (size_t)row * IN;
    const bf16_t* b0B = wt + (size_t)n0 * IN;
    const bf16_t* b1B = wt + (size_t)n1 * IN;
    v8f acc0 = {}, acc1 = {};
    for (int kk = 0; kk < IN; kk += 32) {
        V16bf a, b0, b1;
        a.h[0]  = *(const v8bf*)(aB + kk + hi * 16);
        a.h[1]  = *(const v8bf*)(aB + kk + hi * 16 + 8);
        b0.h[0] = *(const v8bf*)(b0B + kk + hi * 16);
        b0.h[1] = *(const v8bf*)(b0B + kk + hi * 16 + 8);
        b1.h[0] = *(const v8bf*)(b1B + kk + hi * 16);
        b1.h[1] = *(const v8bf*)(b1B + kk + hi * 16 + 8);
        acc0 = WMMA_BF16(a.v, b0.v, acc0);
        acc1 = WMMA_BF16(a.v, b1.v, acc1);
    }
    float bb0 = bias[n0] * LR_, bb1 = bias[n1] * LR_;
    v8bf o0, o1;
    #pragma unroll
    for (int r = 0; r < 8; ++r) {
        float v0 = acc0[r] * wg + bb0; v0 = (v0 > 0.f ? v0 : 0.2f * v0) * SQRT2_;
        float v1 = acc1[r] * wg + bb1; v1 = (v1 > 0.f ? v1 : 0.2f * v1) * SQRT2_;
        o0[r] = (bf16_t)v0; o1[r] = (bf16_t)v1;
    }
    size_t mBase = (size_t)mTile * 16 + hi * 8;              // 8 contiguous rows per lane
    *(v8bf*)(hT + (size_t)n0 * BNP + mBase) = o0;
    *(v8bf*)(hT + (size_t)n1 * BNP + mBase) = o1;
}

// out[b][k][d] = hT[d][b*256], repeated NUM_WS times, f32.
__global__ void gather_kernel(const bf16_t* __restrict__ hT, float* __restrict__ out) {
    int t = blockIdx.x * blockDim.x + threadIdx.x;
    if (t >= B_ * WD) return;
    int b = t / WD, d = t % WD;
    float v = (float)hT[(size_t)d * BNP + b * NP];
    #pragma unroll
    for (int k = 0; k < NUM_WS; ++k)
        out[((size_t)b * NUM_WS + k) * WD + d] = v;
}

// ---------------- host launch ----------------
extern "C" void kernel_launch(void* const* d_in, const int* in_sizes, int n_in,
                              void* d_out, int out_size, void* d_ws, size_t ws_size,
                              hipStream_t stream) {
    (void)in_sizes; (void)n_in; (void)out_size; (void)ws_size;
    const float* z       = (const float*)d_in[0];   // [16,16]
    const float* poses   = (const float*)d_in[1];   // [4000,16]
    const float* adj     = (const float*)d_in[2];   // [4000,4000]
    const float* w_first = (const float*)d_in[3];   // [512,32]
    const float* b_first = (const float*)d_in[4];   // [512]
    const float* w_rest  = (const float*)d_in[5];   // [7,512,512]
    const float* b_rest  = (const float*)d_in[6];   // [7,512]
    float* out = (float*)d_out;                     // [16,14,512]

    // scratch layout
    char* ws = (char*)d_ws;
    const size_t OFF_INV  = 0;                          // 4096 f32           -> 16384
    const size_t OFF_X0T  = OFF_INV  + 16384;           // 32*4096 bf16       -> 262144
    const size_t OFF_ADJP = OFF_X0T  + 262144;          // 16*256*256 bf16    -> 2097152
    const size_t OFF_WT   = OFF_ADJP + 2097152;         // 1851392 bf16       -> 3702784
    const size_t OFF_HT   = OFF_WT   + 3702784;         // 512*4096 bf16      -> 4194304
    const size_t OFF_Y    = OFF_HT   + 4194304;         // 4096*512 bf16      -> 4194304
    float*  inv  = (float*) (ws + OFF_INV);
    bf16_t* x0T  = (bf16_t*)(ws + OFF_X0T);
    bf16_t* adjp = (bf16_t*)(ws + OFF_ADJP);
    bf16_t* wt   = (bf16_t*)(ws + OFF_WT);
    bf16_t* hT   = (bf16_t*)(ws + OFF_HT);
    bf16_t* y    = (bf16_t*)(ws + OFF_Y);

    // prep (single HBM pass over adj; everything else L2-resident afterwards)
    rowsum_kernel<<<BNP / 8, 256, 0, stream>>>(adj, inv);
    adjp_kernel<<<(B_ * NP * NP + 255) / 256, 256, 0, stream>>>(adj, adjp);
    build_x0_kernel<<<(BNP * IN0 + 255) / 256, 256, 0, stream>>>(z, poses, x0T);
    {
        int tot = IN0 * WD + 7 * WD * WD;
        wprep_kernel<<<(tot + 255) / 256, 256, 0, stream>>>(w_first, w_rest, wt);
    }

    // layer 0: agg over x0 (W = 32), then FC 32 -> 512
    agg_kernel<<<dim3(IN0 / 32, 16, B_), 32, 0, stream>>>(adjp, inv, x0T, y, IN0);
    const float wg0 = LR_ / sqrtf((float)IN0);
    fc_kernel<<<dim3(WD / 32, BNP / 16), 32, 0, stream>>>(y, wt, b_first, hT, IN0, wg0);

    // layers 1..7: agg (W = 512), FC 512 -> 512
    const float wgR = LR_ / sqrtf((float)WD);
    for (int l = 1; l < 8; ++l) {
        agg_kernel<<<dim3(WD / 32, 16, B_), 32, 0, stream>>>(adjp, inv, hT, y, WD);
        const bf16_t* wtl = wt + IN0 * WD + (size_t)(l - 1) * WD * WD;
        const float*  bl  = b_rest + (l - 1) * WD;
        fc_kernel<<<dim3(WD / 32, BNP / 16), 32, 0, stream>>>(y, wtl, bl, hT, WD, wgR);
    }

    // gather in_ids rows, repeat NUM_WS, f32 output
    gather_kernel<<<(B_ * WD + 255) / 256, 256, 0, stream>>>(hT, out);
}
